// UNet_TTT_8632884264982
// MI455X (gfx1250) — compile-verified, matrix operand order checked by reading, not run on hardware
//
#include <hip/hip_runtime.h>
#include <hip/hip_bf16.h>

// ---------------------------------------------------------------------------
// Types / helpers
// ---------------------------------------------------------------------------
typedef __bf16 bf16_t;
typedef __attribute__((ext_vector_type(16))) __bf16 v16bf;
typedef __attribute__((ext_vector_type(8)))  __bf16 v8bf;
typedef __attribute__((ext_vector_type(8)))  float  v8f;

__device__ __forceinline__ bf16_t f2bf(float f) {
  unsigned u = __builtin_bit_cast(unsigned, f);
  unsigned r = u + 0x7FFFu + ((u >> 16) & 1u);   // round-to-nearest-even
  unsigned short h = (unsigned short)(r >> 16);
  return __builtin_bit_cast(bf16_t, h);
}
__device__ __forceinline__ bf16_t bf_zero() {
  return __builtin_bit_cast(bf16_t, (unsigned short)0);
}
__device__ __forceinline__ v16bf cat8(v8bf lo, v8bf hi) {
  return __builtin_shufflevector(lo, hi, 0, 1, 2, 3, 4, 5, 6, 7,
                                 8, 9, 10, 11, 12, 13, 14, 15);
}

static inline long cdivl(long a, long b) { return (a + b - 1) / b; }

// ---------------------------------------------------------------------------
// f32 -> bf16 conversion (layout-preserving)
// ---------------------------------------------------------------------------
__global__ void cvt_f32_to_bf16_kernel(const float* __restrict__ s,
                                       bf16_t* __restrict__ d, long n) {
  long i = (long)blockIdx.x * blockDim.x + threadIdx.x;
  if (i < n) d[i] = f2bf(s[i]);
}

// ---------------------------------------------------------------------------
// f32 [K,N] -> bf16 [N,K] transposed conversion (GEMM B operand, N-major so
// per-lane fragment walks are contiguous in K -> global_load_b128 pairs).
// ---------------------------------------------------------------------------
__global__ void cvt_transpose_bf16_kernel(const float* __restrict__ s,
                                          bf16_t* __restrict__ d, int K, int N) {
  long total = (long)K * N;
  long i = (long)blockIdx.x * blockDim.x + threadIdx.x;
  if (i >= total) return;
  int n = (int)(i % N);
  int k = (int)(i / N);
  d[(long)n * K + k] = f2bf(s[i]);
}

// ---------------------------------------------------------------------------
// Conv weights: OIHW f32 -> [Cout][tap=kh*3+kw][Cin] bf16.
// With NHWC activations this makes the implicit-GEMM K walk contiguous.
// ---------------------------------------------------------------------------
__global__ void cvt_w_oihw_to_otc_kernel(const float* __restrict__ s,
                                         bf16_t* __restrict__ d, int Cout, int Cin) {
  long total = (long)Cout * Cin * 9;
  long i = (long)blockIdx.x * blockDim.x + threadIdx.x;
  if (i >= total) return;
  int t = (int)(i % 9);               // s is ((co*Cin+ci)*3+kh)*3+kw = (co*Cin+ci)*9+t
  long r = i / 9;
  int ci = (int)(r % Cin);
  int co = (int)(r / Cin);
  d[((long)co * 9 + t) * Cin + ci] = f2bf(s[i]);
}

// ---------------------------------------------------------------------------
// NCHW f32 -> NHWC f32 (only used once, for the network input)
// ---------------------------------------------------------------------------
__global__ void nchw_to_nhwc_kernel(const float* __restrict__ x,
                                    float* __restrict__ t, int Nb, int C, int HW) {
  long total = (long)Nb * C * HW;
  long i = (long)blockIdx.x * blockDim.x + threadIdx.x;
  if (i >= total) return;
  int hw = (int)(i % HW);
  long r = i / HW;
  int c = (int)(r % C);
  int n = (int)(r / C);
  t[((long)n * HW + hw) * C + c] = x[i];
}

// ---------------------------------------------------------------------------
// bf16 WMMA GEMM, TN form: C[M,N] = A[M,K] x Bt[N,K]^T, f32 out.
// M,N multiples of 16; K multiple of 64. K unrolled x2.
// Fragment packing per CDNA5 ISA 7.12.2 (lanes 0-15: K {0..7,16..23},
// lanes 16-31: K {8..15,24..31}; A rows / B columns in lane%16).
// ---------------------------------------------------------------------------
__global__ __launch_bounds__(128) void gemm_bf16_tn_wmma_kernel(
    const bf16_t* __restrict__ A, const bf16_t* __restrict__ Bt,
    float* __restrict__ C, int M, int N, int K) {
  const int lane = threadIdx.x & 31;
  const int wave = threadIdx.x >> 5;
  const int tn = blockIdx.x * (blockDim.x >> 5) + wave;
  const int tm = blockIdx.y;
  if (tn * 16 >= N) return;
  const int r    = lane & 15;
  const int half = (lane >> 4) & 1;
  const bf16_t* __restrict__ arow = A  + ((long)tm * 16 + r) * K;
  const bf16_t* __restrict__ brow = Bt + ((long)tn * 16 + r) * K;
  v8f acc = {};
  for (int k0 = 0; k0 < K; k0 += 64) {
    __builtin_prefetch(arow + k0 + 64, 0, 0);
    __builtin_prefetch(brow + k0 + 64, 0, 0);
    v8bf a0l = *(const v8bf*)(arow + k0 + half * 8);
    v8bf a0h = *(const v8bf*)(arow + k0 + 16 + half * 8);
    v8bf b0l = *(const v8bf*)(brow + k0 + half * 8);
    v8bf b0h = *(const v8bf*)(brow + k0 + 16 + half * 8);
    v8bf a1l = *(const v8bf*)(arow + k0 + 32 + half * 8);
    v8bf a1h = *(const v8bf*)(arow + k0 + 48 + half * 8);
    v8bf b1l = *(const v8bf*)(brow + k0 + 32 + half * 8);
    v8bf b1h = *(const v8bf*)(brow + k0 + 48 + half * 8);
    acc = __builtin_amdgcn_wmma_f32_16x16x32_bf16(
        false, cat8(a0l, a0h), false, cat8(b0l, b0h), (short)0, acc, false, false);
    acc = __builtin_amdgcn_wmma_f32_16x16x32_bf16(
        false, cat8(a1l, a1h), false, cat8(b1l, b1h), (short)0, acc, false, false);
  }
#pragma unroll
  for (int i = 0; i < 8; ++i) {
    int row = tm * 16 + i + (half << 3);
    C[(long)row * N + tn * 16 + (lane & 15)] = acc[i];
  }
}

// ---------------------------------------------------------------------------
// Implicit-GEMM 3x3 conv (pad=1), NHWC bf16 in / [Cout][9][Cin] bf16 weights /
// NHWC f32 out. Requires Cin % 32 == 0. Per 9 taps the K walk is contiguous
// in Cin, so both operands stream as global_load_b128 pairs; per-lane stores
// are 8 consecutive couts at one pixel (vectorizable).
// Bias omitted: training-mode BatchNorm cancels it exactly.
// ---------------------------------------------------------------------------
__global__ __launch_bounds__(128) void conv3x3_nhwc_wmma_kernel(
    const bf16_t* __restrict__ in, const bf16_t* __restrict__ wgt,
    float* __restrict__ out, int Nb, int Cin, int H, int W, int Cout) {
  const int lane = threadIdx.x & 31;
  const int wave = threadIdx.x >> 5;
  const int HW = H * W;
  const long npix = (long)Nb * HW;
  const long tn = (long)blockIdx.x * (blockDim.x >> 5) + wave;
  const int  tm = blockIdx.y;
  if (tn * 16 >= npix) return;
  const int r    = lane & 15;
  const int half = (lane >> 4) & 1;
  const long p = tn * 16 + r;                 // this lane's pixel (B column)
  const int n  = (int)(p / HW);
  const int hw = (int)(p - (long)n * HW);
  const int ph = hw / W;
  const int pw = hw - ph * W;
  const bf16_t* __restrict__ wrow = wgt + ((long)tm * 16 + r) * 9 * Cin;
  const v8bf vz = {};
  v8f acc = {};
#pragma unroll
  for (int seg = 0; seg < 9; ++seg) {
    const int kh = seg / 3, kw = seg - kh * 3;
    const int ih = ph + kh - 1, iw = pw + kw - 1;
    const bool valid = (ih >= 0) && (ih < H) && (iw >= 0) && (iw < W);
    const long ibase = ((long)(n * H + ih) * W + iw) * Cin;
    const bf16_t* __restrict__ wseg = wrow + seg * Cin;
    for (int c0 = 0; c0 < Cin; c0 += 32) {
      v8bf alo = *(const v8bf*)(wseg + c0 + half * 8);
      v8bf ahi = *(const v8bf*)(wseg + c0 + 16 + half * 8);
      v8bf blo = vz, bhi = vz;
      if (valid) {
        blo = *(const v8bf*)(in + ibase + c0 + half * 8);
        bhi = *(const v8bf*)(in + ibase + c0 + 16 + half * 8);
      }
      acc = __builtin_amdgcn_wmma_f32_16x16x32_bf16(
          false, cat8(alo, ahi), false, cat8(blo, bhi), (short)0, acc, false, false);
    }
  }
  float* __restrict__ orow = out + p * Cout + tm * 16 + half * 8;
#pragma unroll
  for (int i = 0; i < 8; ++i) orow[i] = acc[i];
}

// ---------------------------------------------------------------------------
// Small-Cin fallback (Cin % 32 != 0; only the 3->64 stem conv).
// Scalar im2col gather, K = 9*Cin ordered (tap, ci); NHWC in/out.
// ---------------------------------------------------------------------------
__global__ __launch_bounds__(128) void conv3x3_nhwc_smallc_wmma_kernel(
    const bf16_t* __restrict__ in, const bf16_t* __restrict__ wgt,
    float* __restrict__ out, int Nb, int Cin, int H, int W, int Cout) {
  const int K = Cin * 9;
  const int lane = threadIdx.x & 31;
  const int wave = threadIdx.x >> 5;
  const int HW = H * W;
  const long npix = (long)Nb * HW;
  const long tn = (long)blockIdx.x * (blockDim.x >> 5) + wave;
  const int  tm = blockIdx.y;
  if (tn * 16 >= npix) return;
  const int r    = lane & 15;
  const int half = (lane >> 4) & 1;
  const long p = tn * 16 + r;
  const int n  = (int)(p / HW);
  const int hw = (int)(p - (long)n * HW);
  const int ph = hw / W;
  const int pw = hw - ph * W;
  const bf16_t* __restrict__ wrow = wgt + ((long)tm * 16 + r) * 9 * Cin;
  v8f acc = {};
  for (int k0 = 0; k0 < K; k0 += 32) {
    v16bf af, bfg;
#pragma unroll
    for (int i = 0; i < 16; ++i) {
      int k = k0 + ((i >> 3) << 4) + (half << 3) + (i & 7);
      bf16_t a = bf_zero(), b = bf_zero();
      if (k < K) {
        a = wrow[k];
        int seg = k / Cin;                   // k = seg*Cin + ci
        int ci = k - seg * Cin;
        int kh = seg / 3, kw = seg - kh * 3;
        int ih = ph + kh - 1, iw = pw + kw - 1;
        if (ih >= 0 && ih < H && iw >= 0 && iw < W)
          b = in[((long)(n * H + ih) * W + iw) * Cin + ci];
      }
      af[i] = a;
      bfg[i] = b;
    }
    acc = __builtin_amdgcn_wmma_f32_16x16x32_bf16(
        false, af, false, bfg, (short)0, acc, false, false);
  }
  float* __restrict__ orow = out + p * Cout + tm * 16 + half * 8;
#pragma unroll
  for (int i = 0; i < 8; ++i) orow[i] = acc[i];
}

// ---------------------------------------------------------------------------
// Training-mode BatchNorm over NHWC: per-channel mean / rsqrt(var+eps).
// Strided per-channel reads; the whole tensor is L2-resident (<= 17MB vs
// 192MB L2) so cross-channel cacheline reuse is absorbed by L2.
// ---------------------------------------------------------------------------
__global__ void bn_stats_nhwc_kernel(const float* __restrict__ x,
                                     float* __restrict__ mean,
                                     float* __restrict__ rstd, long P, int C,
                                     float eps) {
  const int c = blockIdx.x;
  __shared__ float s1[256], s2[256];
  float a = 0.f, b = 0.f;
  for (long i = threadIdx.x; i < P; i += blockDim.x) {
    float v = x[i * C + c];
    a += v;
    b += v * v;
  }
  s1[threadIdx.x] = a;
  s2[threadIdx.x] = b;
  __syncthreads();
  for (int s = 128; s > 0; s >>= 1) {
    if ((int)threadIdx.x < s) {
      s1[threadIdx.x] += s1[threadIdx.x + s];
      s2[threadIdx.x] += s2[threadIdx.x + s];
    }
    __syncthreads();
  }
  if (threadIdx.x == 0) {
    float m = s1[0] / (float)P;
    float v = s2[0] / (float)P - m * m;
    mean[c] = m;
    rstd[c] = rsqrtf(v + eps);
  }
}

__global__ void bn_relu_nhwc_kernel(float* __restrict__ x,
                                    const float* __restrict__ mean,
                                    const float* __restrict__ rstd,
                                    const float* __restrict__ g,
                                    const float* __restrict__ be, int C, long total) {
  long i = (long)blockIdx.x * blockDim.x + threadIdx.x;
  if (i >= total) return;
  int c = (int)(i % C);
  float y = (x[i] - mean[c]) * rstd[c] * g[c] + be[c];
  x[i] = y > 0.f ? y : 0.f;
}

// ---------------------------------------------------------------------------
// 2x2 max pool, stride 2, NHWC (channel-fastest -> coalesced)
// ---------------------------------------------------------------------------
__global__ void maxpool2_nhwc_kernel(const float* __restrict__ in,
                                     float* __restrict__ out, int Nb, int C, int H,
                                     int W) {
  const int Ho = H >> 1, Wo = W >> 1;
  long total = (long)Nb * Ho * Wo * C;
  long i = (long)blockIdx.x * blockDim.x + threadIdx.x;
  if (i >= total) return;
  int c = (int)(i % C);
  long t = i / C;
  int wo = (int)(t % Wo);
  t /= Wo;
  int ho = (int)(t % Ho);
  int n = (int)(t / Ho);
  const float* base = in + ((long)(n * H + 2 * ho) * W + 2 * wo) * C + c;
  out[i] = fmaxf(fmaxf(base[0], base[C]),
                 fmaxf(base[(long)W * C], base[(long)(W + 1) * C]));
}

// ---------------------------------------------------------------------------
// LayerNorm over channel dim (NHWC row = token), bf16 output for WMMA GEMMs.
// ---------------------------------------------------------------------------
__global__ __launch_bounds__(64) void layernorm_bf16_kernel(
    const float* __restrict__ t, const float* __restrict__ g,
    const float* __restrict__ be, bf16_t* __restrict__ o, int C) {
  __shared__ float s1[64], s2[64];
  __shared__ float mu, rv;
  const long tok = blockIdx.x;
  const float* row = t + tok * C;
  float a = 0.f, b = 0.f;
  for (int c = threadIdx.x; c < C; c += 64) {
    float x = row[c];
    a += x;
    b += x * x;
  }
  s1[threadIdx.x] = a;
  s2[threadIdx.x] = b;
  __syncthreads();
  for (int s = 32; s > 0; s >>= 1) {
    if ((int)threadIdx.x < s) {
      s1[threadIdx.x] += s1[threadIdx.x + s];
      s2[threadIdx.x] += s2[threadIdx.x + s];
    }
    __syncthreads();
  }
  if (threadIdx.x == 0) {
    mu = s1[0] / (float)C;
    float var = s2[0] / (float)C - mu * mu;
    rv = rsqrtf(var + 1e-5f);
  }
  __syncthreads();
  for (int c = threadIdx.x; c < C; c += 64)
    o[tok * C + c] = f2bf((row[c] - mu) * rv * g[c] + be[c]);
}

// ---------------------------------------------------------------------------
// TTT-Linear inner scan. One block per (batch, head); fast weight (hd x hd)
// LDS-resident across the serial mini-batch loop. Output written as bf16
// directly into the staging buffer for the Wo WMMA GEMM.
// ---------------------------------------------------------------------------
__global__ __launch_bounds__(256) void ttt_scan_kernel(
    const float* __restrict__ q, const float* __restrict__ k,
    const float* __restrict__ v, const float* __restrict__ W1,
    const float* __restrict__ b1, const float* __restrict__ lr,
    bf16_t* __restrict__ zo, int B, int L, int D, int HH) {
  const int hd = D / HH;
  const int bh = blockIdx.x;
  const int b = bh / HH;
  const int h = bh % HH;
  const float eta = lr[0] / (float)hd;
  __shared__ float Ws[64 * 64];
  __shared__ float Xq[16 * 64], Xk[16 * 64], Xv[16 * 64], G[16 * 64], Zb[16 * 64];
  __shared__ float bvec[64];
  __shared__ float attn[16 * 16];
  const int tid = threadIdx.x;
  for (int i = tid; i < hd * hd; i += blockDim.x) Ws[i] = W1[(long)h * hd * hd + i];
  for (int i = tid; i < hd; i += blockDim.x) bvec[i] = b1[h * hd + i];
  __syncthreads();
  const int nc = L / 16;
  for (int c = 0; c < nc; ++c) {
    const long tokBase = (long)b * L + (long)c * 16;
    for (int i = tid; i < 16 * hd; i += blockDim.x) {
      int row = i / hd, d = i - row * hd;
      long off = (tokBase + row) * D + h * hd + d;
      Xq[i] = q[off];
      Xk[i] = k[off];
      Xv[i] = v[off];
    }
    __syncthreads();
    for (int i = tid; i < 16 * hd; i += blockDim.x) {
      int row = i / hd, d = i - row * hd;
      float zk = bvec[d], zq = bvec[d];
      for (int e = 0; e < hd; ++e) {
        float w = Ws[e * hd + d];
        zk += Xk[row * hd + e] * w;
        zq += Xq[row * hd + e] * w;
      }
      G[i] = zk - Xv[i];
      Zb[i] = zq;
    }
    __syncthreads();
    for (int i = tid; i < 256; i += blockDim.x) {
      int row = i >> 4, col = i & 15;
      float s = 0.f;
      if (col <= row)
        for (int e = 0; e < hd; ++e) s += Xq[row * hd + e] * Xk[col * hd + e];
      attn[i] = s;
    }
    __syncthreads();
    for (int i = tid; i < 16 * hd; i += blockDim.x) {
      int row = i / hd, d = i - row * hd;
      float s = 0.f, cs = 0.f;
      for (int j = 0; j <= row; ++j) {
        float gv = G[j * hd + d];
        s += attn[row * 16 + j] * gv;
        cs += gv;
      }
      zo[(tokBase + row) * D + h * hd + d] = f2bf(Zb[i] - eta * (s + cs));
    }
    __syncthreads();
    for (int i = tid; i < hd * hd; i += blockDim.x) {
      int a = i / hd, d = i - a * hd;
      float s = 0.f;
      for (int m = 0; m < 16; ++m) s += Xk[m * hd + a] * G[m * hd + d];
      Ws[i] -= eta * s;
    }
    for (int d = tid; d < hd; d += blockDim.x) {
      float s = 0.f;
      for (int m = 0; m < 16; ++m) s += G[m * hd + d];
      bvec[d] -= eta * s;
    }
    __syncthreads();
  }
}

// ---------------------------------------------------------------------------
// 2x bilinear upsample, align_corners=True, NHWC
// ---------------------------------------------------------------------------
__global__ void upsample2x_nhwc_kernel(const float* __restrict__ in,
                                       float* __restrict__ out, int Nb, int C, int H,
                                       int W) {
  const int Ho = 2 * H, Wo = 2 * W;
  long total = (long)Nb * Ho * Wo * C;
  long i = (long)blockIdx.x * blockDim.x + threadIdx.x;
  if (i >= total) return;
  int c = (int)(i % C);
  long t = i / C;
  int wo = (int)(t % Wo);
  t /= Wo;
  int ho = (int)(t % Ho);
  int n = (int)(t / Ho);
  float fh = (H > 1) ? ho * (float)(H - 1) / (float)(Ho - 1) : 0.f;
  float fw = (W > 1) ? wo * (float)(W - 1) / (float)(Wo - 1) : 0.f;
  int h0 = (int)floorf(fh);
  int h1 = h0 + 1 < H ? h0 + 1 : H - 1;
  int w0 = (int)floorf(fw);
  int w1 = w0 + 1 < W ? w0 + 1 : W - 1;
  float wh = fh - (float)h0, ww = fw - (float)w0;
  const float* base = in + (long)n * H * W * C + c;
  out[i] = base[((long)h0 * W + w0) * C] * (1.f - wh) * (1.f - ww) +
           base[((long)h0 * W + w1) * C] * (1.f - wh) * ww +
           base[((long)h1 * W + w0) * C] * wh * (1.f - ww) +
           base[((long)h1 * W + w1) * C] * wh * ww;
}

// ---------------------------------------------------------------------------
// NHWC channel-slice copy: dst[pix, c0+c] = src[pix, c]
// ---------------------------------------------------------------------------
__global__ void copy_channels_nhwc_kernel(const float* __restrict__ s,
                                          float* __restrict__ d, long P, int Cs,
                                          int Cd, int c0) {
  long total = P * Cs;
  long i = (long)blockIdx.x * blockDim.x + threadIdx.x;
  if (i >= total) return;
  int c = (int)(i % Cs);
  long pix = i / Cs;
  d[pix * Cd + c0 + c] = s[i];
}

// ---------------------------------------------------------------------------
// Final 1x1 conv (64 -> 1) + bias + sigmoid, NHWC (contiguous per-pixel dot).
// For C==1 the NHWC and NCHW flat layouts coincide, matching d_out.
// ---------------------------------------------------------------------------
__global__ void conv1x1_sigmoid_nhwc_kernel(const float* __restrict__ in,
                                            const float* __restrict__ w,
                                            const float* __restrict__ b,
                                            float* __restrict__ out, long P, int C) {
  long i = (long)blockIdx.x * blockDim.x + threadIdx.x;
  if (i >= P) return;
  const float* base = in + i * C;
  float s = b[0];
  for (int c = 0; c < C; ++c) s += base[c] * w[c];
  out[i] = 1.f / (1.f + expf(-s));
}

// ---------------------------------------------------------------------------
// Host orchestration (everything NHWC after the input transpose)
// ---------------------------------------------------------------------------
struct DCp { const float *w1, *g1, *be1, *w2, *g2, *be2; };
struct TTp { const float *ln_g, *ln_b, *Wq, *Wk, *Wv, *Wo, *W1, *b1, *lr; };

extern "C" void kernel_launch(void* const* d_in, const int* in_sizes, int n_in,
                              void* d_out, int out_size, void* d_ws, size_t ws_size,
                              hipStream_t stream) {
  (void)in_sizes; (void)n_in; (void)out_size; (void)ws_size;
  const int B = 4;
  const float* x = (const float*)d_in[0];

  auto getDC = [&](int base) {
    DCp p;
    p.w1 = (const float*)d_in[base + 0];
    p.g1 = (const float*)d_in[base + 2];
    p.be1 = (const float*)d_in[base + 3];
    p.w2 = (const float*)d_in[base + 4];
    p.g2 = (const float*)d_in[base + 6];
    p.be2 = (const float*)d_in[base + 7];
    return p;
  };
  auto getTT = [&](int base) {
    TTp p;
    p.ln_g = (const float*)d_in[base + 0];
    p.ln_b = (const float*)d_in[base + 1];
    p.Wq = (const float*)d_in[base + 2];
    p.Wk = (const float*)d_in[base + 3];
    p.Wv = (const float*)d_in[base + 4];
    p.Wo = (const float*)d_in[base + 5];
    p.W1 = (const float*)d_in[base + 6];
    p.b1 = (const float*)d_in[base + 7];
    p.lr = (const float*)d_in[base + 8];
    return p;
  };
  DCp inc = getDC(1), dn1 = getDC(9), dn2 = getDC(17), dn3 = getDC(25), dn4 = getDC(33);
  TTp tt1 = getTT(41), tt2 = getTT(50), tt3 = getTT(59), tt4 = getTT(68);
  DCp up1 = getDC(77), up2 = getDC(85), up3 = getDC(93), up4 = getDC(101);
  const float* ow = (const float*)d_in[109];
  const float* ob = (const float*)d_in[110];

  // bump allocator over d_ws
  char* wsp = (char*)d_ws;
  size_t off = 0;
  auto alloc = [&](size_t bytes) -> void* {
    size_t a = (off + 255) & ~(size_t)255;
    off = a + bytes;
    return (void*)(wsp + a);
  };
  float* x0  = (float*)alloc(4ull * 3 * 128 * 128 * 4);       // input, NHWC
  float* x1  = (float*)alloc(4ull * 64 * 128 * 128 * 4);
  float* x2  = (float*)alloc(4ull * 128 * 64 * 64 * 4);
  float* x3  = (float*)alloc(4ull * 256 * 32 * 32 * 4);
  float* x4  = (float*)alloc(4ull * 512 * 16 * 16 * 4);
  float* x5  = (float*)alloc(4ull * 512 * 8 * 8 * 4);
  float* x1p = (float*)alloc(4ull * 64 * 128 * 128 * 4);
  float* x2p = (float*)alloc(4ull * 128 * 64 * 64 * 4);
  float* x3p = (float*)alloc(4ull * 256 * 32 * 32 * 4);
  float* x4p = (float*)alloc(4ull * 512 * 16 * 16 * 4);
  float* y4  = (float*)alloc(4ull * 256 * 16 * 16 * 4);
  float* y3  = (float*)alloc(4ull * 128 * 32 * 32 * 4);
  float* y2  = (float*)alloc(4ull * 64 * 64 * 64 * 4);
  float* y1  = (float*)alloc(4ull * 64 * 128 * 128 * 4);
  float* catb  = (float*)alloc(8388608ull * 4);
  float* tmp   = (float*)alloc(4194304ull * 4);
  float* poolb = (float*)alloc(1048576ull * 4);
  float* upb   = (float*)alloc(4194304ull * 4);
  float* tq = (float*)alloc(4194304ull * 4);
  float* tk = (float*)alloc(4194304ull * 4);
  float* tv = (float*)alloc(4194304ull * 4);
  bf16_t* bfA = (bf16_t*)alloc(8388608ull * 2);
  bf16_t* bfW = (bf16_t*)alloc(4194304ull * 2);
  float* mean = (float*)alloc(512 * 4);
  float* rstd = (float*)alloc(512 * 4);

  auto cvt = [&](const float* s, bf16_t* d, long n) {
    cvt_f32_to_bf16_kernel<<<(unsigned)cdivl(n, 256), 256, 0, stream>>>(s, d, n);
  };
  auto cvt_t = [&](const float* s, bf16_t* d, int K, int N) {
    long n = (long)K * N;
    cvt_transpose_bf16_kernel<<<(unsigned)cdivl(n, 256), 256, 0, stream>>>(s, d, K, N);
  };
  auto run_conv = [&](const float* inF, int Nb, int Cin, int H, int W,
                      const float* wF, int Cout, float* outF) {
    cvt(inF, bfA, (long)Nb * H * W * Cin);
    long nW = (long)Cout * Cin * 9;
    cvt_w_oihw_to_otc_kernel<<<(unsigned)cdivl(nW, 256), 256, 0, stream>>>(wF, bfW,
                                                                           Cout, Cin);
    long npix = (long)Nb * H * W;
    dim3 grid((unsigned)cdivl(cdivl(npix, 16), 4), (unsigned)cdivl(Cout, 16));
    if ((Cin & 31) == 0)
      conv3x3_nhwc_wmma_kernel<<<grid, 128, 0, stream>>>(bfA, bfW, outF, Nb, Cin, H, W,
                                                         Cout);
    else
      conv3x3_nhwc_smallc_wmma_kernel<<<grid, 128, 0, stream>>>(bfA, bfW, outF, Nb, Cin,
                                                                H, W, Cout);
  };
  auto run_bn_relu = [&](float* buf, int Nb, int C, int H, int W,
                         const float* g, const float* be) {
    long P = (long)Nb * H * W;
    bn_stats_nhwc_kernel<<<C, 256, 0, stream>>>(buf, mean, rstd, P, C, 1e-5f);
    long total = P * C;
    bn_relu_nhwc_kernel<<<(unsigned)cdivl(total, 256), 256, 0, stream>>>(buf, mean, rstd,
                                                                         g, be, C, total);
  };
  auto run_dc = [&](const float* inF, int Nb, int Cin, int H, int W,
                    const DCp& p, int Cout, float* outF) {
    run_conv(inF, Nb, Cin, H, W, p.w1, Cout, tmp);
    run_bn_relu(tmp, Nb, Cout, H, W, p.g1, p.be1);
    run_conv(tmp, Nb, Cout, H, W, p.w2, Cout, outF);
    run_bn_relu(outF, Nb, Cout, H, W, p.g2, p.be2);
  };
  auto run_gemm = [&](const bf16_t* A, const bf16_t* Bt, float* C, long M, int N, int K) {
    dim3 grid((unsigned)cdivl(cdivl(N, 16), 4), (unsigned)cdivl(M, 16));
    gemm_bf16_tn_wmma_kernel<<<grid, 128, 0, stream>>>(A, Bt, C, (int)M, N, K);
  };
  auto run_ttt = [&](const float* xF, int C, int H, int W, const TTp& p, float* outF) {
    long L = (long)H * W;
    long M = (long)B * L;
    // NHWC buffer *is* the token layout [B*L, C]: no transposes needed.
    layernorm_bf16_kernel<<<(unsigned)M, 64, 0, stream>>>(xF, p.ln_g, p.ln_b, bfA, C);
    cvt_t(p.Wq, bfW, C, C); run_gemm(bfA, bfW, tq, M, C, C);
    cvt_t(p.Wk, bfW, C, C); run_gemm(bfA, bfW, tk, M, C, C);
    cvt_t(p.Wv, bfW, C, C); run_gemm(bfA, bfW, tv, M, C, C);
    // scan consumes q/k/v and overwrites bfA (LN tokens no longer needed)
    ttt_scan_kernel<<<B * 8, 256, 0, stream>>>(tq, tk, tv, p.W1, p.b1, p.lr, bfA,
                                               B, (int)L, C, 8);
    cvt_t(p.Wo, bfW, C, C);
    run_gemm(bfA, bfW, outF, M, C, C);   // output tokens == NHWC activation
  };
  auto run_pool = [&](const float* inF, int C, int H, int W, float* outF) {
    long total = (long)B * (H / 2) * (W / 2) * C;
    maxpool2_nhwc_kernel<<<(unsigned)cdivl(total, 256), 256, 0, stream>>>(inF, outF, B, C,
                                                                          H, W);
  };
  auto run_up = [&](const float* lo, int Cl, int Hl, int Wl, const float* skip, int Cs,
                    const DCp& p, int Cout, float* outF) {
    int Ho = 2 * Hl, Wo = 2 * Wl;
    long P = (long)B * Ho * Wo;
    long tot_up = P * Cl;
    upsample2x_nhwc_kernel<<<(unsigned)cdivl(tot_up, 256), 256, 0, stream>>>(lo, upb, B,
                                                                             Cl, Hl, Wl);
    copy_channels_nhwc_kernel<<<(unsigned)cdivl(P * Cs, 256), 256, 0, stream>>>(
        skip, catb, P, Cs, Cs + Cl, 0);
    copy_channels_nhwc_kernel<<<(unsigned)cdivl(tot_up, 256), 256, 0, stream>>>(
        upb, catb, P, Cl, Cs + Cl, Cs);
    run_dc(catb, B, Cs + Cl, Ho, Wo, p, Cout, outF);
  };

  // ---- input to NHWC ----
  {
    long total = 4ull * 3 * 128 * 128;
    nchw_to_nhwc_kernel<<<(unsigned)cdivl(total, 256), 256, 0, stream>>>(x, x0, B, 3,
                                                                         128 * 128);
  }

  // ---- encoder ----
  run_dc(x0, B, 3, 128, 128, inc, 64, x1);
  run_pool(x1, 64, 128, 128, poolb);  run_dc(poolb, B, 64, 64, 64, dn1, 128, x2);
  run_pool(x2, 128, 64, 64, poolb);   run_dc(poolb, B, 128, 32, 32, dn2, 256, x3);
  run_pool(x3, 256, 32, 32, poolb);   run_dc(poolb, B, 256, 16, 16, dn3, 512, x4);
  run_pool(x4, 512, 16, 16, poolb);   run_dc(poolb, B, 512, 8, 8, dn4, 512, x5);

  // ---- TTT skip transforms ----
  run_ttt(x1, 64, 128, 128, tt1, x1p);
  run_ttt(x2, 128, 64, 64, tt2, x2p);
  run_ttt(x3, 256, 32, 32, tt3, x3p);
  run_ttt(x4, 512, 16, 16, tt4, x4p);

  // ---- decoder ----
  run_up(x5, 512, 8, 8,   x4p, 512, up1, 256, y4);
  run_up(y4, 256, 16, 16, x3p, 256, up2, 128, y3);
  run_up(y3, 128, 32, 32, x2p, 128, up3, 64, y2);
  run_up(y2, 64, 64, 64,  x1p, 64,  up4, 64, y1);

  // ---- head ----
  long opix = (long)B * 128 * 128;
  conv1x1_sigmoid_nhwc_kernel<<<(unsigned)cdivl(opix, 256), 256, 0, stream>>>(
      y1, ow, ob, (float*)d_out, opix, 64);
}
